// CustomTransformerEncoder_70205535421169
// MI455X (gfx1250) — compile-verified, matrix operand order checked by reading, not run on hardware
//
#include <hip/hip_runtime.h>

// ---------------------------------------------------------------------------
// Types / WMMA helpers (CDNA5 gfx1250, wave32)
// ---------------------------------------------------------------------------
typedef __attribute__((ext_vector_type(16))) __bf16 v16bf;
typedef __attribute__((ext_vector_type(8)))  float  v8f;

union V8   { v8f  v; float f[8]; };
union Frag { v16bf v; uint4 u[2]; };

// Load one 16x16x32 bf16 fragment (16 bf16 per lane) from p (first 8 K-vals)
// and p+16 (second 8 K-vals), matching the ISA 16-bit A/B VGPR layout.
__device__ inline v16bf load_frag(const __bf16* p) {
    Frag f;
    f.u[0] = *(const uint4*)(p);
    f.u[1] = *(const uint4*)(p + 16);
    return f.v;
}

__device__ inline v8f wmma_bf16(v16bf a, v16bf b, v8f c) {
    return __builtin_amdgcn_wmma_f32_16x16x32_bf16(
        /*neg_a=*/false, a, /*neg_b=*/false, b,
        /*c_mod=*/(short)0, c, /*reuse_a=*/false, /*reuse_b=*/false);
}

// ---- CDNA5 async global->LDS copy (ASYNCcnt-tracked, no VGPR round trip) ---
// VDST VGPR carries the wave-relative LDS byte address (ISA 15.18 async ops).
__device__ inline unsigned lds_off_of(const void* p) {
    return (unsigned)(unsigned long long)
        (__attribute__((address_space(3))) const char*)p;
}
__device__ inline void async_ld_b128(unsigned lds_byte_off, const void* g) {
    asm volatile("global_load_async_to_lds_b128 %0, %1, off"
                 :: "v"(lds_byte_off), "v"((unsigned long long)g)
                 : "memory");
}
__device__ inline void wait_async0() {
    asm volatile("s_wait_asynccnt 0" ::: "memory");
}

// ---------------------------------------------------------------------------
// Problem constants (match reference)
// ---------------------------------------------------------------------------
constexpr int kB     = 2;
constexpr int kS     = 2304;     // 2048 long + 256 short
constexpr int kLong  = 2048;
constexpr int kD     = 1024;
constexpr int kH     = 16;
constexpr int kDh    = 64;
constexpr int kF     = 1024;
constexpr int kL     = 4;
constexpr int kM     = kB * kS;  // 4608 rows

// ---------------------------------------------------------------------------
// Elementwise kernels
// ---------------------------------------------------------------------------
__global__ void cvt_f32_bf16_kernel(const float* __restrict__ in,
                                    __bf16* __restrict__ out, size_t n) {
    size_t i = (size_t)blockIdx.x * blockDim.x + threadIdx.x;
    if (i < n) out[i] = (__bf16)in[i];
}

__global__ void bias_relu_cvt_kernel(const float* __restrict__ in,
                                     const float* __restrict__ bias,
                                     __bf16* __restrict__ out,
                                     size_t n, int ncols) {
    size_t i = (size_t)blockIdx.x * blockDim.x + threadIdx.x;
    if (i < n) {
        float v = in[i] + bias[i % ncols];
        out[i] = (__bf16)fmaxf(v, 0.0f);
    }
}

// qkv[B,S,3D] f32 (+bias) -> q,k [B,H,S,dh] bf16 (q pre-scaled by 1/sqrt(dh)),
// v transposed to [B,H,dh,S] bf16 so attention B-fragments are contiguous.
__global__ void split_qkv_kernel(const float* __restrict__ qkv,
                                 const float* __restrict__ bias,
                                 __bf16* __restrict__ q,
                                 __bf16* __restrict__ k,
                                 __bf16* __restrict__ vt) {
    size_t i = (size_t)blockIdx.x * blockDim.x + threadIdx.x;
    size_t n = (size_t)kM * kD;
    if (i >= n) return;
    int d = (int)(i % kD);
    size_t bs = i / kD;
    int s = (int)(bs % kS);
    int b = (int)(bs / kS);
    int h = d / kDh, dd = d % kDh;
    const float* base = qkv + bs * (3 * kD);
    float qv = (base[d]          + bias[d])          * 0.125f; // 1/sqrt(64)
    float kv =  base[kD + d]     + bias[kD + d];
    float vv =  base[2 * kD + d] + bias[2 * kD + d];
    size_t bh = (size_t)(b * kH + h);
    q [(bh * kS + s) * kDh + dd] = (__bf16)qv;
    k [(bh * kS + s) * kDh + dd] = (__bf16)kv;
    vt[(bh * kDh + dd) * kS + s] = (__bf16)vv;
}

// out = LN(a + r + (bias?bias:0)) * w + beta ; one 256-thread block per row.
__global__ __launch_bounds__(256)
void residual_ln_kernel(const float* __restrict__ a,
                        const float* __restrict__ r,
                        const float* __restrict__ bias,   // may be null
                        const float* __restrict__ w,
                        const float* __restrict__ beta,
                        float* __restrict__ out_f32,
                        __bf16* __restrict__ out_bf) {    // may be null
    __shared__ float red1[8];
    __shared__ float red2[8];
    int row = blockIdx.x;
    int t = threadIdx.x;
    const float* ap = a + (size_t)row * kD;
    const float* rp = r + (size_t)row * kD;

    float vals[4];
    float s = 0.0f;
#pragma unroll
    for (int j = 0; j < 4; ++j) {
        int d = t + j * 256;
        float v = ap[d] + rp[d] + (bias ? bias[d] : 0.0f);
        vals[j] = v;
        s += v;
    }
#pragma unroll
    for (int o = 16; o; o >>= 1) s += __shfl_xor(s, o);
    if ((t & 31) == 0) red1[t >> 5] = s;
    __syncthreads();
    float tot = 0.0f;
#pragma unroll
    for (int i = 0; i < 8; ++i) tot += red1[i];
    float mean = tot * (1.0f / kD);

    float vs = 0.0f;
#pragma unroll
    for (int j = 0; j < 4; ++j) {
        float dvi = vals[j] - mean;
        vs += dvi * dvi;
    }
#pragma unroll
    for (int o = 16; o; o >>= 1) vs += __shfl_xor(vs, o);
    if ((t & 31) == 0) red2[t >> 5] = vs;
    __syncthreads();
    float vtot = 0.0f;
#pragma unroll
    for (int i = 0; i < 8; ++i) vtot += red2[i];
    float inv = rsqrtf(vtot * (1.0f / kD) + 1e-5f);

#pragma unroll
    for (int j = 0; j < 4; ++j) {
        int d = t + j * 256;
        float o = (vals[j] - mean) * inv * w[d] + beta[d];
        out_f32[(size_t)row * kD + d] = o;
        if (out_bf) out_bf[(size_t)row * kD + d] = (__bf16)o;
    }
}

// ---------------------------------------------------------------------------
// Tiled WMMA GEMM: C[M,N] f32 = A[M,K]bf16 * B[N,K]bf16^T
// Block tile 128x128x32, 8 waves, each wave 64x32 (4x2 WMMA accumulators).
// K-tiles stream through a double-buffered LDS stage filled with
// global_load_async_to_lds_b128; one s_wait_asynccnt + one barrier per iter.
// ---------------------------------------------------------------------------
constexpr int TM = 128, TN = 128, TK = 32, LDSS = 40; // padded LDS stride

__global__ __launch_bounds__(256)
void gemm_bf16_wmma_kernel(const __bf16* __restrict__ A,
                           const __bf16* __restrict__ B,
                           float* __restrict__ C,
                           int K, int lda, int ldb, int ldc) {
    __shared__ __bf16 As[2][TM * LDSS];
    __shared__ __bf16 Bs[2][TN * LDSS];

    int t    = threadIdx.x;
    int lane = t & 31, wid = t >> 5;
    int g    = lane >> 4;        // lane half (K-split select)
    int lr   = lane & 15;
    int mblk = blockIdx.y * TM;
    int nblk = blockIdx.x * TN;
    int wm   = (wid & 1) * 64;   // wave M offset within block tile
    int wn   = (wid >> 1) * 32;  // wave N offset

    int row_ld = t >> 1;         // 0..127, cooperative tile load
    int col_ld = (t & 1) * 16;   // 0 or 16

    // Issue this thread's 4x16B async copies for K-tile k0 into buffer `buf`.
    auto issue_tile = [&](int buf, int k0) {
        const __bf16* ga = A + (size_t)(mblk + row_ld) * lda + k0 + col_ld;
        const __bf16* gb = B + (size_t)(nblk + row_ld) * ldb + k0 + col_ld;
        unsigned la = lds_off_of(&As[buf][row_ld * LDSS + col_ld]);
        unsigned lb = lds_off_of(&Bs[buf][row_ld * LDSS + col_ld]);
        async_ld_b128(la,      ga);
        async_ld_b128(la + 16, ga + 8);
        async_ld_b128(lb,      gb);
        async_ld_b128(lb + 16, gb + 8);
    };

    V8 acc[4][2];
#pragma unroll
    for (int i = 0; i < 4; ++i)
#pragma unroll
        for (int j = 0; j < 2; ++j)
#pragma unroll
            for (int rr = 0; rr < 8; ++rr) acc[i][j].f[rr] = 0.0f;

    // prologue: stage tile 0
    issue_tile(0, 0);
    wait_async0();
    __syncthreads();

    int nIter = K / TK;
    for (int it = 0; it < nIter; ++it) {
        int cur = it & 1;
        if (it + 1 < nIter) issue_tile(cur ^ 1, (it + 1) * TK);

        v16bf af[4], bfg[2];
#pragma unroll
        for (int i = 0; i < 4; ++i)
            af[i] = load_frag(&As[cur][(wm + i * 16 + lr) * LDSS + g * 8]);
#pragma unroll
        for (int j = 0; j < 2; ++j)
            bfg[j] = load_frag(&Bs[cur][(wn + j * 16 + lr) * LDSS + g * 8]);
#pragma unroll
        for (int i = 0; i < 4; ++i)
#pragma unroll
            for (int j = 0; j < 2; ++j)
                acc[i][j].v = wmma_bf16(af[i], bfg[j], acc[i][j].v);

        // my next-tile async writes done + my reads done -> one barrier
        wait_async0();
        __syncthreads();
    }

#pragma unroll
    for (int i = 0; i < 4; ++i) {
#pragma unroll
        for (int j = 0; j < 2; ++j) {
            int row0 = mblk + wm + i * 16 + g * 8;   // C layout: row = g*8 + r
            int col  = nblk + wn + j * 16 + lr;      // col = lane&15
#pragma unroll
            for (int rr = 0; rr < 8; ++rr)
                C[(size_t)(row0 + rr) * ldc + col] = acc[i][j].f[rr];
        }
    }
}

// ---------------------------------------------------------------------------
// Flash attention with band+diagonal mask.
// Q,K: [B*H, S, 64] bf16 (Q pre-scaled). Vt: [B*H, 64, S] bf16.
// Output ctx written as bf16 in [B, S, D] (heads re-interleaved).
// 4 waves/block, each wave owns 16 query rows.
// ---------------------------------------------------------------------------
__global__ __launch_bounds__(128)
void flash_attn_kernel(const __bf16* __restrict__ Q,
                       const __bf16* __restrict__ Km,
                       const __bf16* __restrict__ Vt,
                       __bf16* __restrict__ ctx) {
    __shared__ __bf16 Pt[4][16 * 32];   // per-wave P tile (row-major 16x32)

    int lane = threadIdx.x & 31, wid = threadIdx.x >> 5;
    int g = lane >> 4, lr = lane & 15;
    int bh = blockIdx.y;
    int qb = (blockIdx.x * 4 + wid) * 16;

    const __bf16* qbase = Q  + ((size_t)bh * kS + qb) * kDh;
    const __bf16* kbase = Km + ((size_t)bh * kS) * kDh;
    const __bf16* vbase = Vt + ((size_t)bh * kDh) * kS;
    __bf16* Pw = Pt[wid];

    // Q fragments: rows qb+lr, dh split into two K=32 fragments
    v16bf qf0 = load_frag(qbase + (size_t)lr * kDh + 0  + g * 8);
    v16bf qf1 = load_frag(qbase + (size_t)lr * kDh + 32 + g * 8);

    float mrow[8], lrow[8];
    V8 c[4];
#pragma unroll
    for (int rr = 0; rr < 8; ++rr) { mrow[rr] = -3.0e38f; lrow[rr] = 0.0f; }
#pragma unroll
    for (int tI = 0; tI < 4; ++tI)
#pragma unroll
        for (int rr = 0; rr < 8; ++rr) c[tI].f[rr] = 0.0f;

    const bool shortBlk = (qb >= kLong);
    const int nch = kLong / 32;                 // 64 long-key chunks

    for (int ci = 0; ci <= nch; ++ci) {
        if (ci == nch && !shortBlk) break;
        const bool diag = (ci == nch);
        const int kc = diag ? (qb & ~31) : ci * 32;

        // ---- scores: two 16x16 tiles over 32 keys ----
        V8 s0, s1;
#pragma unroll
        for (int rr = 0; rr < 8; ++rr) { s0.f[rr] = 0.0f; s1.f[rr] = 0.0f; }
        v16bf kf00 = load_frag(kbase + (size_t)(kc + lr)      * kDh + 0  + g * 8);
        v16bf kf01 = load_frag(kbase + (size_t)(kc + lr)      * kDh + 32 + g * 8);
        v16bf kf10 = load_frag(kbase + (size_t)(kc + 16 + lr) * kDh + 0  + g * 8);
        v16bf kf11 = load_frag(kbase + (size_t)(kc + 16 + lr) * kDh + 32 + g * 8);
        s0.v = wmma_bf16(qf0, kf00, s0.v);
        s0.v = wmma_bf16(qf1, kf01, s0.v);
        s1.v = wmma_bf16(qf0, kf10, s1.v);
        s1.v = wmma_bf16(qf1, kf11, s1.v);

        if (diag) {
#pragma unroll
            for (int rr = 0; rr < 8; ++rr) {
                int rowq = qb + g * 8 + rr;
                if (kc + lr      != rowq) s0.f[rr] = -1.0e30f;
                if (kc + 16 + lr != rowq) s1.f[rr] = -1.0e30f;
            }
        }

        // ---- streaming softmax update (row = g*8 + r, shared by 16 lanes) --
        float scale[8];
#pragma unroll
        for (int rr = 0; rr < 8; ++rr) {
            float a = fmaxf(s0.f[rr], s1.f[rr]);
#pragma unroll
            for (int o = 8; o; o >>= 1) a = fmaxf(a, __shfl_xor(a, o));
            float mnew = fmaxf(mrow[rr], a);
            float sc = __expf(mrow[rr] - mnew);
            float p0 = __expf(s0.f[rr] - mnew);
            float p1 = __expf(s1.f[rr] - mnew);
            float ps = p0 + p1;
#pragma unroll
            for (int o = 8; o; o >>= 1) ps += __shfl_xor(ps, o);
            lrow[rr] = lrow[rr] * sc + ps;
            mrow[rr] = mnew;
            scale[rr] = sc;
            Pw[(g * 8 + rr) * 32 + lr]      = (__bf16)p0;
            Pw[(g * 8 + rr) * 32 + 16 + lr] = (__bf16)p1;
        }
        asm volatile("s_wait_dscnt 0" ::: "memory");   // P stores visible

#pragma unroll
        for (int tI = 0; tI < 4; ++tI)
#pragma unroll
            for (int rr = 0; rr < 8; ++rr) c[tI].f[rr] *= scale[rr];

        // ---- ctx += P(16x32) * V(32x64): A-frag of P from LDS ----
        v16bf pf = load_frag(Pw + lr * 32 + g * 8);
#pragma unroll
        for (int tI = 0; tI < 4; ++tI) {
            v16bf vf = load_frag(vbase + (size_t)(tI * 16 + lr) * kS + kc + g * 8);
            c[tI].v = wmma_bf16(pf, vf, c[tI].v);
        }
    }

    // ---- epilogue: normalize, re-interleave heads into [B,S,D] bf16 ----
    int b = bh / kH, h = bh % kH;
#pragma unroll
    for (int rr = 0; rr < 8; ++rr) {
        float inv = 1.0f / lrow[rr];
        int srow = qb + g * 8 + rr;
        __bf16* op = ctx + ((size_t)(b * kS + srow)) * kD + h * kDh;
#pragma unroll
        for (int tI = 0; tI < 4; ++tI)
            op[tI * 16 + lr] = (__bf16)(c[tI].f[rr] * inv);
    }
}

// ---------------------------------------------------------------------------
// Host orchestration
// ---------------------------------------------------------------------------
static inline unsigned ceil_div(size_t n, unsigned b) {
    return (unsigned)((n + b - 1) / b);
}

extern "C" void kernel_launch(void* const* d_in, const int* in_sizes, int n_in,
                              void* d_out, int out_size, void* d_ws, size_t ws_size,
                              hipStream_t stream) {
    (void)in_sizes; (void)n_in; (void)out_size; (void)ws_size;

    const float* x_in  = (const float*)d_in[0];
    const float* Wqkv  = (const float*)d_in[1];
    const float* bqkv  = (const float*)d_in[2];
    const float* Wo    = (const float*)d_in[3];
    const float* bo    = (const float*)d_in[4];
    const float* W1    = (const float*)d_in[5];
    const float* b1    = (const float*)d_in[6];
    const float* W2    = (const float*)d_in[7];
    const float* b2    = (const float*)d_in[8];
    const float* ln1w  = (const float*)d_in[9];
    const float* ln1b  = (const float*)d_in[10];
    const float* ln2w  = (const float*)d_in[11];
    const float* ln2b  = (const float*)d_in[12];
    const float* normw = (const float*)d_in[13];
    const float* normb = (const float*)d_in[14];

    // bump allocator over d_ws
    size_t off = 0;
    auto alloc = [&](size_t bytes) -> void* {
        off = (off + 255) & ~(size_t)255;
        void* p = (char*)d_ws + off;
        off += bytes;
        return p;
    };
    const size_t MD = (size_t)kM * kD;
    __bf16* WqkvBF = (__bf16*)alloc((size_t)kL * 3 * kD * kD * 2);
    __bf16* WoBF   = (__bf16*)alloc((size_t)kL * kD * kD * 2);
    __bf16* W1BF   = (__bf16*)alloc((size_t)kL * kF * kD * 2);
    __bf16* W2BF   = (__bf16*)alloc((size_t)kL * kD * kF * 2);
    float*  x0     = (float*) alloc(MD * 4);
    float*  x1     = (float*) alloc(MD * 4);
    __bf16* x_bf   = (__bf16*)alloc(MD * 2);
    float*  qkv    = (float*) alloc(MD * 3 * 4);
    __bf16* q_bf   = (__bf16*)alloc(MD * 2);
    __bf16* k_bf   = (__bf16*)alloc(MD * 2);
    __bf16* vt_bf  = (__bf16*)alloc(MD * 2);
    __bf16* ctx_bf = (__bf16*)alloc(MD * 2);
    float*  attn   = (float*) alloc(MD * 4);
    float*  h_f32  = (float*) alloc(MD * 4);
    __bf16* h_bf   = (__bf16*)alloc(MD * 2);
    float*  ff1    = (float*) alloc((size_t)kM * kF * 4);
    __bf16* ff1_bf = (__bf16*)alloc((size_t)kM * kF * 2);
    float*  ff2    = (float*) alloc(MD * 4);
    float*  y_f32  = (float*) alloc(MD * 4);

    // weight conversion fp32 -> bf16 (once per launch; deterministic)
    {
        size_t n;
        n = (size_t)kL * 3 * kD * kD;
        cvt_f32_bf16_kernel<<<ceil_div(n, 256), 256, 0, stream>>>(Wqkv, WqkvBF, n);
        n = (size_t)kL * kD * kD;
        cvt_f32_bf16_kernel<<<ceil_div(n, 256), 256, 0, stream>>>(Wo, WoBF, n);
        n = (size_t)kL * kF * kD;
        cvt_f32_bf16_kernel<<<ceil_div(n, 256), 256, 0, stream>>>(W1, W1BF, n);
        n = (size_t)kL * kD * kF;
        cvt_f32_bf16_kernel<<<ceil_div(n, 256), 256, 0, stream>>>(W2, W2BF, n);
    }

    auto gemm = [&](const __bf16* A, const __bf16* B, float* C,
                    int M, int N, int K, int lda, int ldb, int ldc) {
        dim3 grid(N / TN, M / TM);
        gemm_bf16_wmma_kernel<<<grid, 256, 0, stream>>>(A, B, C, K, lda, ldb, ldc);
    };

    float* xbuf[2] = { x0, x1 };
    for (int l = 0; l < kL; ++l) {
        const float* xin  = (l == 0) ? x_in : xbuf[(l + 1) & 1];
        float*       xout = (l == kL - 1) ? (float*)d_out : xbuf[l & 1];

        // x -> bf16
        cvt_f32_bf16_kernel<<<ceil_div(MD, 256), 256, 0, stream>>>(xin, x_bf, MD);

        // QKV projection: [4608,1024] x [3072,1024]^T
        gemm(x_bf, WqkvBF + (size_t)l * 3 * kD * kD, qkv,
             kM, 3 * kD, kD, kD, kD, 3 * kD);

        // split into heads (+bias, q scaled, v transposed)
        split_qkv_kernel<<<ceil_div(MD, 256), 256, 0, stream>>>(
            qkv, bqkv + (size_t)l * 3 * kD, q_bf, k_bf, vt_bf);

        // flash attention with band+diag mask
        flash_attn_kernel<<<dim3(kS / 64, kB * kH), 128, 0, stream>>>(
            q_bf, k_bf, vt_bf, ctx_bf);

        // output projection
        gemm(ctx_bf, WoBF + (size_t)l * kD * kD, attn, kM, kD, kD, kD, kD, kD);

        // h = LN(x + attn + bo)
        residual_ln_kernel<<<kM, 256, 0, stream>>>(
            xin, attn, bo + (size_t)l * kD,
            ln1w + (size_t)l * kD, ln1b + (size_t)l * kD, h_f32, h_bf);

        // FF1 (+bias+relu+cvt)
        gemm(h_bf, W1BF + (size_t)l * kF * kD, ff1, kM, kF, kD, kD, kD, kF);
        bias_relu_cvt_kernel<<<ceil_div((size_t)kM * kF, 256), 256, 0, stream>>>(
            ff1, b1 + (size_t)l * kF, ff1_bf, (size_t)kM * kF, kF);

        // FF2
        gemm(ff1_bf, W2BF + (size_t)l * kD * kF, ff2, kM, kD, kF, kF, kF, kD);

        // y = LN(h + ff2 + b2)
        residual_ln_kernel<<<kM, 256, 0, stream>>>(
            h_f32, ff2, b2 + (size_t)l * kD,
            ln2w + (size_t)l * kD, ln2b + (size_t)l * kD, y_f32, nullptr);

        // x_next = LN(x + y) with shared norm
        residual_ln_kernel<<<kM, 256, 0, stream>>>(
            xin, y_f32, nullptr, normw, normb, xout, nullptr);
    }
}
